// LSTMDecoder_24026047054250
// MI455X (gfx1250) — compile-verified, matrix operand order checked by reading, not run on hardware
//
#include <hip/hip_runtime.h>

#define HDIM 256
#define TM   64          // batch rows per workgroup
#define MT   (TM / 16)   // M-tiles per column stripe
#define BATCH 65536
#define NTHR 256         // 8 wave32s

typedef __attribute__((ext_vector_type(16))) __bf16 v16bf;
typedef __attribute__((ext_vector_type(8)))  float  v8f;
typedef __attribute__((ext_vector_type(4)))  float  v4f;

// ---------------------------------------------------------------------------
// Explicit global-address-space accessors: force global_load/global_store
// (flat loads would tie LOADcnt to DScnt and serialize against LDS traffic).
// ---------------------------------------------------------------------------
#define AS1 __attribute__((address_space(1)))
__device__ __forceinline__ unsigned ldg_u32(const void* p) {
    return *(const AS1 unsigned*)(unsigned long long)p;
}
__device__ __forceinline__ float ldg_f32(const float* p) {
    return *(const AS1 float*)(unsigned long long)p;
}
__device__ __forceinline__ v4f ldg_f32x4(const float* p) {
    return *(const AS1 v4f*)(unsigned long long)p;
}
__device__ __forceinline__ void stg_f32(float* p, float v) {
    *(AS1 float*)(unsigned long long)p = v;
}

// ---------------------------------------------------------------------------
// WMMA fragment helpers (CDNA5 ISA 7.12.2 16-bit layouts, wave32)
// A (16x32): lane L holds row M=L%16; K = (elem<8?0:16) + (L>=16?8:0) + elem%8
// B (32x16): symmetric, lane L holds col N=L%16 with same K striping
// C (16x16 f32): lanes 0-15: N=L, M=r; lanes 16-31: N=L-16, M=r+8
// ---------------------------------------------------------------------------
__device__ __forceinline__ int kmap(int lane, int p) {
    return ((p & 4) << 2) + ((lane & 16) >> 1) + ((p & 3) << 1);
}

__device__ __forceinline__ v16bf load_tileA(const __bf16* buf, int mb, int kb, int lane) {
    union { unsigned u[8]; v16bf v; } r;
    const unsigned short* p =
        (const unsigned short*)(buf + (mb + (lane & 15)) * HDIM + kb);
#pragma unroll
    for (int i = 0; i < 8; ++i)
        r.u[i] = *(const unsigned*)(p + kmap(lane, i));   // LDS: ds_load
    return r.v;
}

__device__ __forceinline__ v16bf load_tileB(const __bf16* __restrict__ W, int nbase,
                                            int kb, int ks, int lane) {
    union { unsigned u[8]; v16bf v; } r;
    const unsigned short* p =
        (const unsigned short*)(W + (size_t)(nbase + (lane & 15)) * ks + kb);
#pragma unroll
    for (int i = 0; i < 8; ++i)
        r.u[i] = ldg_u32(p + kmap(lane, i));              // global_load_b128 clauses
    return r.v;
}

__device__ __forceinline__ v8f wmma_bf16(v16bf a, v16bf b, v8f c) {
    return __builtin_amdgcn_wmma_f32_16x16x32_bf16(false, a, false, b, (short)0, c,
                                                   false, false);
}

// Lean activations: v_exp_f32 + v_rcp_f32 (~1 ulp), correct +/-inf saturation.
// Avoids the fully-guarded OCML tanhf (large VALU/branch footprint per call).
__device__ __forceinline__ float fast_sigmoid(float x) {
    return __builtin_amdgcn_rcpf(1.0f + __expf(-x));
}
__device__ __forceinline__ float fast_tanh(float x) {
    return 1.0f - 2.0f * __builtin_amdgcn_rcpf(__expf(2.0f * x) + 1.0f);
}

// ---------------------------------------------------------------------------
// LSTM cell stage. Waves own 16-wide N column stripes (2 per wave). Per K-step:
// A tiles for all MT row blocks loaded once (reused by 4 gates), each gate's B
// fragment loaded once (reused by MT row blocks). acc[4][MT] stays in VGPRs.
// ---------------------------------------------------------------------------
__device__ void lstm_stage(const __bf16* __restrict__ Wih,   // [1024,256] bf16
                           const __bf16* __restrict__ Whh,   // [1024,256] or null
                           const float*  __restrict__ gb,    // [1024] = bih+bhh
                           const __bf16* xb, const __bf16* hprev, const __bf16* cprev,
                           __bf16* hout, __bf16* cout, int lane, int wid) {
    for (int nt = wid; nt < 16; nt += 8) {
        const int nb = nt << 4;
        v8f z = {};
        v8f acc[4][MT];
#pragma unroll
        for (int g = 0; g < 4; ++g)
#pragma unroll
            for (int mt = 0; mt < MT; ++mt) acc[g][mt] = z;

        for (int kb = 0; kb < HDIM; kb += 32) {
            v16bf a[MT];
#pragma unroll
            for (int mt = 0; mt < MT; ++mt)
                a[mt] = load_tileA(xb, mt << 4, kb, lane);
#pragma unroll
            for (int g = 0; g < 4; ++g) {
                v16bf b = load_tileB(Wih, g * HDIM + nb, kb, HDIM, lane);
#pragma unroll
                for (int mt = 0; mt < MT; ++mt)
                    acc[g][mt] = wmma_bf16(a[mt], b, acc[g][mt]);
            }
        }
        if (Whh) {
            for (int kb = 0; kb < HDIM; kb += 32) {
                v16bf a[MT];
#pragma unroll
                for (int mt = 0; mt < MT; ++mt)
                    a[mt] = load_tileA(hprev, mt << 4, kb, lane);
#pragma unroll
                for (int g = 0; g < 4; ++g) {
                    v16bf b = load_tileB(Whh, g * HDIM + nb, kb, HDIM, lane);
#pragma unroll
                    for (int mt = 0; mt < MT; ++mt)
                        acc[g][mt] = wmma_bf16(a[mt], b, acc[g][mt]);
                }
            }
        }
        const int n    = nb + (lane & 15);
        const int mofs = (lane & 16) ? 8 : 0;
        const float bi  = ldg_f32(gb + 0 * HDIM + n);
        const float bf_ = ldg_f32(gb + 1 * HDIM + n);
        const float bg  = ldg_f32(gb + 2 * HDIM + n);
        const float bo  = ldg_f32(gb + 3 * HDIM + n);
#pragma unroll
        for (int mt = 0; mt < MT; ++mt) {
#pragma unroll
            for (int r = 0; r < 8; ++r) {
                const int m = (mt << 4) + r + mofs;
                float iv = fast_sigmoid(acc[0][mt][r] + bi);
                float fv = fast_sigmoid(acc[1][mt][r] + bf_);
                float gv = fast_tanh(acc[2][mt][r] + bg);
                float ov = fast_sigmoid(acc[3][mt][r] + bo);
                float c2 = iv * gv;
                if (cprev) c2 += fv * (float)cprev[m * HDIM + n];
                float h2 = ov * fast_tanh(c2);
                hout[m * HDIM + n] = (__bf16)h2;
                if (cout) cout[m * HDIM + n] = (__bf16)c2;
            }
        }
    }
}

// out = src @ W^T + b + addb   (state mixing for hin2 / cin2)
__device__ void mix_stage(const __bf16* src, const __bf16* __restrict__ W,
                          const float* __restrict__ bv, const __bf16* addb,
                          __bf16* out, int lane, int wid) {
    for (int nt = wid; nt < 16; nt += 8) {
        const int nb = nt << 4;
        v8f z = {};
        v8f acc[MT];
#pragma unroll
        for (int mt = 0; mt < MT; ++mt) acc[mt] = z;
        for (int kb = 0; kb < HDIM; kb += 32) {
            v16bf b = load_tileB(W, nb, kb, HDIM, lane);
#pragma unroll
            for (int mt = 0; mt < MT; ++mt)
                acc[mt] = wmma_bf16(load_tileA(src, mt << 4, kb, lane), b, acc[mt]);
        }
        const int n = nb + (lane & 15);
        const int mofs = (lane & 16) ? 8 : 0;
        const float b = ldg_f32(bv + n);
#pragma unroll
        for (int mt = 0; mt < MT; ++mt)
#pragma unroll
            for (int r = 0; r < 8; ++r) {
                const int m = (mt << 4) + r + mofs;
                out[m * HDIM + n] = (__bf16)(acc[mt][r] + b + (float)addb[m * HDIM + n]);
            }
    }
}

// m = relu(h @ midW^T + midb)
__device__ void mid_stage(const __bf16* hsrc, const __bf16* __restrict__ W,
                          const float* __restrict__ bv, __bf16* mout,
                          int lane, int wid) {
    for (int nt = wid; nt < 16; nt += 8) {
        const int nb = nt << 4;
        v8f z = {};
        v8f acc[MT];
#pragma unroll
        for (int mt = 0; mt < MT; ++mt) acc[mt] = z;
        for (int kb = 0; kb < HDIM; kb += 32) {
            v16bf b = load_tileB(W, nb, kb, HDIM, lane);
#pragma unroll
            for (int mt = 0; mt < MT; ++mt)
                acc[mt] = wmma_bf16(load_tileA(hsrc, mt << 4, kb, lane), b, acc[mt]);
        }
        const int n = nb + (lane & 15);
        const int mofs = (lane & 16) ? 8 : 0;
        const float b = ldg_f32(bv + n);
#pragma unroll
        for (int mt = 0; mt < MT; ++mt)
#pragma unroll
            for (int r = 0; r < 8; ++r) {
                const int m = (mt << 4) + r + mofs;
                mout[m * HDIM + n] = (__bf16)fmaxf(acc[mt][r] + b, 0.0f);
            }
    }
}

// out = m @ outW_pad^T + outb  (outW zero-padded to padN rows); store n < NC only
// Heads are tiny: keep per-tile scheduling for wave load balance.
__device__ void out_stage(const __bf16* msrc, const __bf16* __restrict__ Wo,
                          const float* __restrict__ bv, float* __restrict__ dst,
                          int NC, int padN, int rowbase, int lane, int wid) {
    const int nt = padN / 16;
    const int UN = (TM / 16) * nt;
    for (int unit = wid; unit < UN; unit += 8) {
        const int mb = (unit / nt) * 16;
        const int nb = (unit % nt) * 16;
        v8f c = {};
#pragma unroll
        for (int kb = 0; kb < HDIM; kb += 32)
            c = wmma_bf16(load_tileA(msrc, mb, kb, lane),
                          load_tileB(Wo, nb, kb, HDIM, lane), c);
        const int n = nb + (lane & 15);
        if (n < NC) {
            const int mofs = (lane & 16) ? 8 : 0;
            const float b = ldg_f32(bv + n);
#pragma unroll
            for (int r = 0; r < 8; ++r) {
                const int m = rowbase + mb + r + mofs;
                stg_f32(dst + (size_t)m * NC + n, c[r] + b);
            }
        }
    }
}

// ---------------------------------------------------------------------------
// Fused decoder: whole live dataflow per 64-row batch tile, states in LDS
// (8 x 64x256 bf16 buffers = 256 KB of the 320 KB WGP LDS).
// ---------------------------------------------------------------------------
__global__ __launch_bounds__(NTHR)
void fused_lstm_decoder(const float* __restrict__ x,
                        const __bf16* __restrict__ wih,    // [3][1024][256]
                        const __bf16* __restrict__ whh12,  // Whh[1],Whh[2]
                        const __bf16* __restrict__ hsw,    // hsW[1][2]
                        const __bf16* __restrict__ csw,    // csW[1][2]
                        const __bf16* __restrict__ midw,   // [3][256][256]
                        const __bf16* __restrict__ ow0,    // padded [64][256]
                        const __bf16* __restrict__ ow1,    // padded [32][256]
                        const __bf16* __restrict__ ow2,    // padded [32][256]
                        const float* __restrict__ gbias,   // [3][1024] bih+bhh
                        const float* __restrict__ hsb12, const float* __restrict__ csb12,
                        const float* __restrict__ midb,
                        const float* __restrict__ ob0, const float* __restrict__ ob1,
                        const float* __restrict__ ob2,
                        float* __restrict__ out) {
    extern __shared__ __bf16 smem[];
    __bf16* xb   = smem + 0 * TM * HDIM;
    __bf16* h0   = smem + 1 * TM * HDIM;
    __bf16* c0   = smem + 2 * TM * HDIM;
    __bf16* h1   = smem + 3 * TM * HDIM;
    __bf16* c1   = smem + 4 * TM * HDIM;
    __bf16* hin2 = smem + 5 * TM * HDIM;
    __bf16* cin2 = smem + 6 * TM * HDIM;
    __bf16* h2   = smem + 7 * TM * HDIM;
    // c0/c1/cin2 are dead after the task-2 cell; reuse as the three mid buffers
    __bf16* m0 = c0;
    __bf16* m1 = c1;
    __bf16* m2 = cin2;

    const int tid = threadIdx.x, lane = tid & 31, wid = tid >> 5;
    const int rowbase = blockIdx.x * TM;
    const int WSZ = 4 * HDIM * HDIM;     // one [1024,256] weight

    // Stage x tile as bf16 via b128 global loads + merged ds stores
    for (int i = tid; i < TM * HDIM / 4; i += NTHR) {
        v4f v = ldg_f32x4(x + (size_t)rowbase * HDIM + (size_t)i * 4);
        __bf16* d = xb + i * 4;
        d[0] = (__bf16)v.x; d[1] = (__bf16)v.y;
        d[2] = (__bf16)v.z; d[3] = (__bf16)v.w;
    }
    __syncthreads();

    // task 0 cell (h=c=0)
    lstm_stage(wih, nullptr, gbias, xb, nullptr, nullptr, h0, c0, lane, wid);
    __syncthreads();
    // task 1 cell
    lstm_stage(wih + WSZ, whh12, gbias + 4 * HDIM, xb, h0, c0, h1, c1, lane, wid);
    __syncthreads();
    // state mixing feeding task 2
    mix_stage(h1, hsw, hsb12, h0, hin2, lane, wid);
    mix_stage(c1, csw, csb12, c0, cin2, lane, wid);
    __syncthreads();
    // task 2 cell (c2 never stored; consumed in-register)
    lstm_stage(wih + 2 * WSZ, whh12 + WSZ, gbias + 8 * HDIM, xb, hin2, cin2,
               h2, nullptr, lane, wid);
    __syncthreads();

    // heads: all three midfc GEMMs, one barrier, then all three output GEMMs
    mid_stage(h0, midw,                   midb,            m0, lane, wid);
    mid_stage(h1, midw + HDIM * HDIM,     midb + HDIM,     m1, lane, wid);
    mid_stage(h2, midw + 2 * HDIM * HDIM, midb + 2 * HDIM, m2, lane, wid);
    __syncthreads();
    out_stage(m0, ow0, ob0, out,                       54, 64, rowbase, lane, wid);
    out_stage(m1, ow1, ob1, out + (size_t)BATCH * 54,  23, 32, rowbase, lane, wid);
    out_stage(m2, ow2, ob2, out + (size_t)BATCH * 77,  20, 32, rowbase, lane, wid);
}

// ---------------------------------------------------------------------------
// Weight prep kernels (f32 -> bf16 into workspace; L2-resident afterwards)
// ---------------------------------------------------------------------------
__global__ void k_cvt(const float* __restrict__ s, __bf16* __restrict__ d, int n) {
    int i = blockIdx.x * blockDim.x + threadIdx.x;
    if (i < n) d[i] = (__bf16)s[i];
}
__global__ void k_cvt_pad(const float* __restrict__ s, __bf16* __restrict__ d,
                          int rows, int cols, int prows) {
    int i = blockIdx.x * blockDim.x + threadIdx.x;
    if (i < prows * cols) {
        int r = i / cols;
        d[i] = (r < rows) ? (__bf16)s[i] : (__bf16)0.0f;
    }
}
__global__ void k_addf(const float* __restrict__ a, const float* __restrict__ b,
                       float* __restrict__ d, int n) {
    int i = blockIdx.x * blockDim.x + threadIdx.x;
    if (i < n) d[i] = a[i] + b[i];
}

extern "C" void kernel_launch(void* const* d_in, const int* in_sizes, int n_in,
                              void* d_out, int out_size, void* d_ws, size_t ws_size,
                              hipStream_t stream) {
    const float* x     = (const float*)d_in[0];
    const float* Wih   = (const float*)d_in[1];
    const float* Whh   = (const float*)d_in[2];
    const float* bih   = (const float*)d_in[3];
    const float* bhh   = (const float*)d_in[4];
    const float* hsW   = (const float*)d_in[5];
    const float* hsb   = (const float*)d_in[6];
    const float* csW   = (const float*)d_in[7];
    const float* csb   = (const float*)d_in[8];
    const float* midW  = (const float*)d_in[9];
    const float* midb  = (const float*)d_in[10];
    const float* outW0 = (const float*)d_in[11];
    const float* outb0 = (const float*)d_in[12];
    const float* outW1 = (const float*)d_in[13];
    const float* outb1 = (const float*)d_in[14];
    const float* outW2 = (const float*)d_in[15];
    const float* outb2 = (const float*)d_in[16];

    char* ws = (char*)d_ws;
    const size_t nWih = (size_t)3 * 1024 * 256;
    const size_t nWhh = (size_t)2 * 1024 * 256;
    const size_t nSq  = (size_t)256 * 256;
    size_t o = 0;
    __bf16* wihb  = (__bf16*)(ws + o); o += nWih * 2;
    __bf16* whhb  = (__bf16*)(ws + o); o += nWhh * 2;
    __bf16* hswb  = (__bf16*)(ws + o); o += nSq * 2;
    __bf16* cswb  = (__bf16*)(ws + o); o += nSq * 2;
    __bf16* midwb = (__bf16*)(ws + o); o += 3 * nSq * 2;
    __bf16* ow0b  = (__bf16*)(ws + o); o += (size_t)64 * 256 * 2;
    __bf16* ow1b  = (__bf16*)(ws + o); o += (size_t)32 * 256 * 2;
    __bf16* ow2b  = (__bf16*)(ws + o); o += (size_t)32 * 256 * 2;
    float*  gb    = (float*)(ws + o);  o += (size_t)3 * 1024 * 4;

    auto blocks = [](size_t n) { return (unsigned)((n + 255) / 256); };

    k_cvt<<<blocks(nWih), 256, 0, stream>>>(Wih, wihb, (int)nWih);
    k_cvt<<<blocks(nWhh), 256, 0, stream>>>(Whh + 1024 * 256, whhb, (int)nWhh);
    k_cvt<<<blocks(nSq), 256, 0, stream>>>(hsW + (size_t)5 * nSq, hswb, (int)nSq);
    k_cvt<<<blocks(nSq), 256, 0, stream>>>(csW + (size_t)5 * nSq, cswb, (int)nSq);
    k_cvt<<<blocks(3 * nSq), 256, 0, stream>>>(midW, midwb, (int)(3 * nSq));
    k_cvt_pad<<<blocks(64 * 256), 256, 0, stream>>>(outW0, ow0b, 54, 256, 64);
    k_cvt_pad<<<blocks(32 * 256), 256, 0, stream>>>(outW1, ow1b, 23, 256, 32);
    k_cvt_pad<<<blocks(32 * 256), 256, 0, stream>>>(outW2, ow2b, 20, 256, 32);
    k_addf<<<blocks(3 * 1024), 256, 0, stream>>>(bih, bhh, gb, 3 * 1024);

    const size_t smem_bytes = (size_t)8 * TM * HDIM * sizeof(__bf16); // 256 KB LDS
    fused_lstm_decoder<<<BATCH / TM, NTHR, smem_bytes, stream>>>(
        x, wihb, whhb, hswb, cswb, midwb, ow0b, ow1b, ow2b,
        gb, hsb + 5 * 256, csb + 5 * 256, midb, outb0, outb1, outb2,
        (float*)d_out);
}
